// RelPartialLearnableMultiHeadAttn_36507222016072
// MI455X (gfx1250) — matrix operand order repaired and checked by reading, not compiled, not run on hardware
//
#include <hip/hip_runtime.h>

// ---------------------------------------------------------------------------
// Transformer-XL decoder layer for MI455X (gfx1250, wave32, WMMA + TDM).
// Compute-bound (~86 GFLOP vs ~10us of HBM time at 23.3 TB/s): all GEMM work
// runs through v_wmma_f32_16x16x32_bf16 (f32 accumulate). Projection GEMMs
// stage tiles into LDS with the Tensor Data Mover (tensor_load_to_lds,
// TENSORcnt); attention GEMMs load fragments straight from global (b128).
// Softmax / layernorm / residuals stay fp32.
// ---------------------------------------------------------------------------

#define DEC 1024
#define BSZ 8
#define ENCL 1024
#define DMODEL 512
#define NHEADS 8
#define DHEAD 64

typedef __attribute__((ext_vector_type(16))) __bf16 v16bf;
typedef __attribute__((ext_vector_type(8))) float v8f;
typedef __attribute__((ext_vector_type(4))) unsigned int u32x4;
typedef __attribute__((ext_vector_type(8))) int i32x8;
typedef __attribute__((ext_vector_type(4))) int i32x4;

__device__ __forceinline__ unsigned short f2bf(float f) {
  unsigned u = __float_as_uint(f);
  u += 0x7fffu + ((u >> 16) & 1u);   // round-to-nearest-even
  return (unsigned short)(u >> 16);
}

union FragB16 { v16bf v; unsigned u[8]; uint4 q[2]; };

// A-matrix fragment (16x32 bf16, MxK). Per ISA 7.12.2:
// lanes 0-15: rows M=0..15, V0..3 = K 0..7,  V4..7 = K 16..23
// lanes 16-31: same rows,   V0..3 = K 8..15, V4..7 = K 24..31
__device__ __forceinline__ v16bf load_frag_a(const unsigned short* __restrict__ base,
                                             int row0, int k0, int ldk, int lane) {
  FragB16 f;
  const int r  = row0 + (lane & 15);
  const int hk = (lane & 16) ? 8 : 0;
  const unsigned short* p = base + (size_t)r * ldk + k0 + hk;
  f.q[0] = *(const uint4*)(p);        // K: hk .. hk+7
  f.q[1] = *(const uint4*)(p + 16);   // K: 16+hk .. 16+hk+7
  return f.v;
}

// B-matrix fragment (32x16 bf16, KxN) from W stored row-major [N][K]
// (B[k][n] = W[n][k]). Lanes 0-15: col N=lane, K 0..7,16..23; lanes 16-31: K 8..15,24..31.
__device__ __forceinline__ v16bf load_frag_b(const unsigned short* __restrict__ base,
                                             int col0, int k0, int ldk, int lane) {
  FragB16 f;
  const int c  = col0 + (lane & 15);
  const int hk = (lane & 16) ? 8 : 0;
  const unsigned short* p = base + (size_t)c * ldk + k0 + hk;
  f.q[0] = *(const uint4*)(p);        // K: hk .. hk+7
  f.q[1] = *(const uint4*)(p + 16);   // K: 16+hk .. 16+hk+7
  return f.v;
}

// Tensor Data Mover: 2D tile (rows x cols bf16, row stride ld elements) from
// global memory to LDS at byte offset lds_byte. D# per ISA ch.8:
//   group0: count=1 | lds_addr | global_addr[56:0] | type=2
//   group1: data_size=1(2B), tensor_dim0=cols, tensor_dim1=rows,
//           tile_dim0=cols, tile_dim1=rows, tensor_dim0_stride=ld
__device__ __forceinline__ void tdm_load_2d(unsigned lds_byte, const void* gptr,
                                            int rows, int cols, int ld) {
  const unsigned long long ga = (unsigned long long)(size_t)gptr;
  u32x4 g0;
  g0[0] = 1u;                                               // count=1 (valid)
  g0[1] = lds_byte;
  g0[2] = (unsigned)(ga & 0xffffffffu);
  g0[3] = (unsigned)((ga >> 32) & 0x01ffffffu) | (2u << 30); // type=2 (image)
  i32x8 g1;
  g1[0] = (int)(1u << 16);                                  // data_size=2B
  g1[1] = (int)(((unsigned)cols & 0xffffu) << 16);          // tensor_dim0 lo16
  g1[2] = (int)((((unsigned)cols >> 16) & 0xffffu) | (((unsigned)rows & 0xffffu) << 16));
  g1[3] = (int)((((unsigned)rows >> 16) & 0xffffu) | (((unsigned)cols & 0xffffu) << 16)); // tile_dim0
  g1[4] = (int)((unsigned)rows & 0xffffu);                  // tile_dim1, tile_dim2=0
  g1[5] = (int)(unsigned)ld;                                // tensor_dim0_stride lo32
  g1[6] = 0;
  g1[7] = 0;
  i32x4 z4 = {0, 0, 0, 0};
#if defined(__clang_major__) && __clang_major__ >= 23
  i32x8 z8 = {};
  __builtin_amdgcn_tensor_load_to_lds(g0, g1, z4, z4, z8, 0);
#else
  __builtin_amdgcn_tensor_load_to_lds(g0, g1, z4, z4, 0);
#endif
}

enum {
  MODE_PLAIN = 0,        // Cf[z][row][col] = acc
  MODE_SELF_SCORE = 1,   // Cf = (acc + BD[z][row][DEC-1+col-row]) * scale, causal mask
  MODE_CROSS_SCORE = 2,  // Cf = acc * scale + (pad ? 0 : -1e30)
  MODE_HEADVEC = 3,      // Ch[(row*BSZ+b)*DMODEL + z*DHEAD + col] = bf16(acc)
  MODE_SPLIT_HEAD = 4    // q-proj scatter into per-head [h][i][d] bf16 layout
};

// C[M,N] (+epilogue) = A[M,K] @ W[N,K]^T, bf16 in, f32 accumulate.
// Block: 128 threads = 4 waves (2x2). Wave tile = (WM*16) x (WN*16).
// USE_LDS: double-buffered TDM staging of A/B tiles into LDS (wave 0 issues).
template <int WM, int WN, bool USE_LDS>
__global__ __launch_bounds__(128)
void gemm_bf16_t(const unsigned short* __restrict__ A,
                 const unsigned short* __restrict__ W,
                 int M, int N, int K,
                 long aBatch, long wBatch, long cBatch,
                 float* __restrict__ Cf,
                 unsigned short* __restrict__ Ch,
                 const float* __restrict__ BD, long bdBatch,
                 const int* __restrict__ pad,
                 float scale, int bIdx, int mode)
{
  (void)M;
  const int z = blockIdx.z;
  A += (size_t)z * aBatch;
  W += (size_t)z * wBatch;
  const int lane = threadIdx.x & 31;
  const int wave = threadIdx.x >> 5;
  const int waveM = wave >> 1, waveN = wave & 1;
  const int m0 = blockIdx.x * (32 * WM) + waveM * (WM * 16);
  const int n0 = blockIdx.y * (32 * WN) + waveN * (WN * 16);

  v8f acc[WM][WN] = {};

  if constexpr (!USE_LDS) {
    for (int k0 = 0; k0 < K; k0 += 32) {
      v16bf af[WM], bfr[WN];
#pragma unroll
      for (int i = 0; i < WM; ++i) af[i] = load_frag_a(A, m0 + 16 * i, k0, K, lane);
#pragma unroll
      for (int j = 0; j < WN; ++j) bfr[j] = load_frag_b(W, n0 + 16 * j, k0, K, lane);
#pragma unroll
      for (int i = 0; i < WM; ++i)
#pragma unroll
        for (int j = 0; j < WN; ++j)
          acc[i][j] = __builtin_amdgcn_wmma_f32_16x16x32_bf16(
              false, af[i], false, bfr[j], (short)0, acc[i][j], false, false);
    }
  } else {
    constexpr int AR = 32 * WM;                 // block-tile rows of A
    constexpr int BR = 32 * WN;                 // block-tile rows of W (cols of C)
    constexpr int TBE = (AR + BR) * 32;         // elements per k-step buffer
    __shared__ unsigned short smem[2 * TBE];
    const int bm = blockIdx.x * AR;
    const int bn = blockIdx.y * BR;
    const bool issuer = (wave == 0);

    auto stage = [&](int k0, int buf) {
      if (issuer) {
        const unsigned base = (unsigned)(buf * TBE * 2);    // LDS byte offset
        tdm_load_2d(base,               A + (size_t)bm * K + k0, AR, 32, K);
        tdm_load_2d(base + AR * 32 * 2, W + (size_t)bn * K + k0, BR, 32, K);
      }
    };

    stage(0, 0);
    if (issuer) __builtin_amdgcn_s_wait_tensorcnt(0);
    __syncthreads();
    const int T = K / 32;
    for (int t = 0; t < T; ++t) {
      const int cur = t & 1;
      if (t + 1 < T) stage(t + 1, cur ^ 1);     // TDM overlaps the WMMAs below
      const unsigned short* sA = smem + cur * TBE;
      const unsigned short* sB = sA + AR * 32;
      v16bf af[WM], bfr[WN];
#pragma unroll
      for (int i = 0; i < WM; ++i)
        af[i] = load_frag_a(sA, waveM * (WM * 16) + 16 * i, 0, 32, lane);
#pragma unroll
      for (int j = 0; j < WN; ++j)
        bfr[j] = load_frag_b(sB, waveN * (WN * 16) + 16 * j, 0, 32, lane);
#pragma unroll
      for (int i = 0; i < WM; ++i)
#pragma unroll
        for (int j = 0; j < WN; ++j)
          acc[i][j] = __builtin_amdgcn_wmma_f32_16x16x32_bf16(
              false, af[i], false, bfr[j], (short)0, acc[i][j], false, false);
      if (issuer) __builtin_amdgcn_s_wait_tensorcnt(0);
      __syncthreads();                           // tile consumed; safe to overwrite
    }
  }

  // C/D layout: VGPR e -> M = e + 8*(lane>=16), N = lane&15  (ISA 7.12.2)
  const int lhalf = (lane >> 4) & 1;
  const int lcol = lane & 15;
#pragma unroll
  for (int ti = 0; ti < WM; ++ti)
#pragma unroll
    for (int tj = 0; tj < WN; ++tj) {
      const int tm = m0 + 16 * ti;
      const int tn = n0 + 16 * tj;
      const v8f av = acc[ti][tj];
#pragma unroll
      for (int e = 0; e < 8; ++e) {
        const int row = tm + e + 8 * lhalf;
        const int col = tn + lcol;
        float v = av[e];
        if (mode == MODE_PLAIN) {
          Cf[(size_t)z * cBatch + (size_t)row * N + col] = v;
        } else if (mode == MODE_SELF_SCORE) {
          float out;
          if (col > row) {
            out = -1e30f;                        // causal mask (tril)
          } else {
            const float bd = BD[(size_t)z * bdBatch + (size_t)row * DEC + (DEC - 1 + col - row)];
            out = (v + bd) * scale;              // rel_shift applied as gather
          }
          Cf[(size_t)z * cBatch + (size_t)row * N + col] = out;
        } else if (mode == MODE_CROSS_SCORE) {
          float out = v * scale;
          if (pad[col * BSZ + bIdx] == 0) out += -1e30f;
          Cf[(size_t)z * cBatch + (size_t)row * N + col] = out;
        } else if (mode == MODE_HEADVEC) {
          // row = i, col = d, z = head n -> attn_vec[(i*BSZ+b)*DMODEL + n*64 + d]
          Ch[(size_t)(row * BSZ + bIdx) * DMODEL + z * DHEAD + col] = f2bf(v);
        } else { // MODE_SPLIT_HEAD: row = i*BSZ+b, col = n*64+d -> [h][i][d]
          const int bb = row & (BSZ - 1), ii = row >> 3;
          const int nn = col >> 6, dd = col & 63;
          Ch[((size_t)(bb * NHEADS + nn) * DEC + ii) * DHEAD + dd] = f2bf(v);
        }
      }
    }
}

__global__ void cvt_bf16_kernel(const float* __restrict__ s, unsigned short* __restrict__ d, int n) {
  int i = blockIdx.x * blockDim.x + threadIdx.x;
  if (i < n) d[i] = f2bf(s[i]);
}

// gbuf fp32 [8192][1536] -> per-head bf16 Q(+r_w_bias), Q(+r_r_bias), K, V^T
__global__ void split_qkv_kernel(const float* __restrict__ g,
                                 const float* __restrict__ rwb, const float* __restrict__ rrb,
                                 unsigned short* __restrict__ Qrw, unsigned short* __restrict__ Qrr,
                                 unsigned short* __restrict__ Kh, unsigned short* __restrict__ VT) {
  const int idx = blockIdx.x * blockDim.x + threadIdx.x;
  const int m = idx >> 9, c = idx & 511;
  const int i = m >> 3, b = m & 7, n = c >> 6, d = c & 63;
  const size_t base = (size_t)(b * NHEADS + n) * (DEC * DHEAD);
  const float* row = g + (size_t)m * (3 * DMODEL);
  const float q = row[c];
  Qrw[base + (size_t)i * DHEAD + d] = f2bf(q + rwb[c]);
  Qrr[base + (size_t)i * DHEAD + d] = f2bf(q + rrb[c]);
  Kh [base + (size_t)i * DHEAD + d] = f2bf(row[DMODEL + c]);
  VT [base + (size_t)d * DEC + i]   = f2bf(row[2 * DMODEL + c]);
}

// gbuf fp32 [1024][512] -> rh bf16 [n][p][d]
__global__ void split_r_kernel(const float* __restrict__ g, unsigned short* __restrict__ rh) {
  const int idx = blockIdx.x * blockDim.x + threadIdx.x;
  const int p = idx >> 9, c = idx & 511;
  const int n = c >> 6, d = c & 63;
  rh[((size_t)n * DEC + p) * DHEAD + d] = f2bf(g[(size_t)p * DMODEL + c]);
}

// gbuf fp32 [8192][1024] (enc KV) -> per-head bf16 K, V^T
__global__ void split_kv_kernel(const float* __restrict__ g,
                                unsigned short* __restrict__ Kh, unsigned short* __restrict__ VT) {
  const int idx = blockIdx.x * blockDim.x + threadIdx.x;
  const int m = idx >> 9, c = idx & 511;
  const int kpos = m >> 3, b = m & 7, n = c >> 6, d = c & 63;
  const size_t base = (size_t)(b * NHEADS + n) * (ENCL * DHEAD);
  const float* row = g + (size_t)m * (2 * DMODEL);
  Kh[base + (size_t)kpos * DHEAD + d] = f2bf(row[c]);
  VT[base + (size_t)d * ENCL + kpos]  = f2bf(row[DMODEL + c]);
}

// Wave-per-row softmax over 1024 cols, fp32 in -> bf16 probabilities.
__global__ void softmax_rows(const float* __restrict__ S, unsigned short* __restrict__ P, int ncols) {
  const int lane = threadIdx.x & 31, wave = threadIdx.x >> 5;
  const int row = blockIdx.x * (blockDim.x >> 5) + wave;
  const float* src = S + (size_t)row * ncols;
  float m = -1e38f;
  for (int j = lane; j < ncols; j += 32) m = fmaxf(m, src[j]);
  for (int o = 16; o; o >>= 1) m = fmaxf(m, __shfl_xor(m, o, 32));
  float s = 0.f;
  for (int j = lane; j < ncols; j += 32) s += __expf(src[j] - m);
  for (int o = 16; o; o >>= 1) s += __shfl_xor(s, o, 32);
  const float inv = 1.0f / s;
  unsigned short* dst = P + (size_t)row * ncols;
  for (int j = lane; j < ncols; j += 32) dst[j] = f2bf(__expf(src[j] - m) * inv);
}

// out = LayerNorm(x + y) * g + b ; optional fp32 and bf16 outputs. Wave per row.
__global__ void add_layernorm_kernel(const float* __restrict__ x, const float* __restrict__ y,
                                     const float* __restrict__ g, const float* __restrict__ b,
                                     float* __restrict__ outf, unsigned short* __restrict__ outh) {
  const int lane = threadIdx.x & 31, wave = threadIdx.x >> 5;
  const int row = blockIdx.x * (blockDim.x >> 5) + wave;
  const float* xr = x + (size_t)row * DMODEL;
  const float* yr = y + (size_t)row * DMODEL;
  float vals[16];
  float mu = 0.f;
#pragma unroll
  for (int t = 0; t < 16; ++t) { const int c = lane + 32 * t; vals[t] = xr[c] + yr[c]; mu += vals[t]; }
  for (int o = 16; o; o >>= 1) mu += __shfl_xor(mu, o, 32);
  mu *= (1.0f / DMODEL);
  float var = 0.f;
#pragma unroll
  for (int t = 0; t < 16; ++t) { const float dv = vals[t] - mu; var += dv * dv; }
  for (int o = 16; o; o >>= 1) var += __shfl_xor(var, o, 32);
  var *= (1.0f / DMODEL);
  const float rstd = rsqrtf(var + 1e-5f);
#pragma unroll
  for (int t = 0; t < 16; ++t) {
    const int c = lane + 32 * t;
    const float o = (vals[t] - mu) * rstd * g[c] + b[c];
    if (outf) outf[(size_t)row * DMODEL + c] = o;
    if (outh) outh[(size_t)row * DMODEL + c] = f2bf(o);
  }
}

extern "C" void kernel_launch(void* const* d_in, const int* in_sizes, int n_in,
                              void* d_out, int out_size, void* d_ws, size_t ws_size,
                              hipStream_t stream) {
  (void)in_sizes; (void)n_in; (void)out_size; (void)ws_size;
  const float* w_f   = (const float*)d_in[0];
  const float* r_f   = (const float*)d_in[1];
  const float* enc_f = (const float*)d_in[2];
  const int*   pad   = (const int*)d_in[4];      // (ENC, BSZ) int32
  const float* qkvW  = (const float*)d_in[5];
  const float* oW    = (const float*)d_in[6];
  const float* rnetW = (const float*)d_in[7];
  const float* kvW   = (const float*)d_in[8];
  const float* qW    = (const float*)d_in[9];
  const float* ioW   = (const float*)d_in[10];
  const float* rwb   = (const float*)d_in[11];
  const float* rrb   = (const float*)d_in[12];
  const float* ln1g  = (const float*)d_in[13];
  const float* ln1b  = (const float*)d_in[14];
  const float* ln2g  = (const float*)d_in[15];
  const float* ln2b  = (const float*)d_in[16];

  char* p = (char*)d_ws;
  auto alloc = [&](size_t bytes) { char* r = p; p += (bytes + 255) & ~(size_t)255; return r; };
  const size_t HEAD = (size_t)DEC * DHEAD;          // 65536 elems per head
  unsigned short* Xw   = (unsigned short*)alloc((size_t)DEC*BSZ*DMODEL*2);
  unsigned short* Xe   = (unsigned short*)alloc((size_t)ENCL*BSZ*DMODEL*2);
  unsigned short* Rb   = (unsigned short*)alloc((size_t)DEC*DMODEL*2);
  unsigned short* Wqkv = (unsigned short*)alloc((size_t)3*DMODEL*DMODEL*2);
  unsigned short* Wrn  = (unsigned short*)alloc((size_t)DMODEL*DMODEL*2);
  unsigned short* Wo   = (unsigned short*)alloc((size_t)DMODEL*DMODEL*2);
  unsigned short* Wq   = (unsigned short*)alloc((size_t)DMODEL*DMODEL*2);
  unsigned short* Wio  = (unsigned short*)alloc((size_t)DMODEL*DMODEL*2);
  unsigned short* Wkv  = (unsigned short*)alloc((size_t)2*DMODEL*DMODEL*2);
  unsigned short* rh   = (unsigned short*)alloc((size_t)NHEADS*HEAD*2);
  unsigned short* Qrw  = (unsigned short*)alloc((size_t)BSZ*NHEADS*HEAD*2);
  unsigned short* Qrr  = (unsigned short*)alloc((size_t)BSZ*NHEADS*HEAD*2);
  unsigned short* Kh   = (unsigned short*)alloc((size_t)BSZ*NHEADS*HEAD*2);
  unsigned short* VT   = (unsigned short*)alloc((size_t)BSZ*NHEADS*HEAD*2);
  unsigned short* AV   = (unsigned short*)alloc((size_t)DEC*BSZ*DMODEL*2);
  unsigned short* XV   = (unsigned short*)alloc((size_t)DEC*BSZ*DMODEL*2);
  float*          ln1f = (float*)alloc((size_t)DEC*BSZ*DMODEL*4);
  unsigned short* ln1h = (unsigned short*)alloc((size_t)DEC*BSZ*DMODEL*2);
  float*          BDf  = (float*)alloc((size_t)NHEADS*DEC*DEC*4);      // per-b
  float*          Sf   = (float*)alloc((size_t)NHEADS*DEC*DEC*4);      // per-b
  unsigned short* Ph   = (unsigned short*)alloc((size_t)NHEADS*DEC*DEC*2);
  float*          gbuf = (float*)alloc((size_t)DEC*BSZ*3*DMODEL*4);    // largest GEMM out

  auto cvt = [&](const float* s, unsigned short* d, int n) {
    cvt_bf16_kernel<<<dim3((n + 255) / 256), dim3(256), 0, stream>>>(s, d, n);
  };
  cvt(w_f,   Xw,   DEC*BSZ*DMODEL);
  cvt(enc_f, Xe,   ENCL*BSZ*DMODEL);
  cvt(r_f,   Rb,   DEC*DMODEL);
  cvt(qkvW,  Wqkv, 3*DMODEL*DMODEL);
  cvt(rnetW, Wrn,  DMODEL*DMODEL);
  cvt(oW,    Wo,   DMODEL*DMODEL);
  cvt(qW,    Wq,   DMODEL*DMODEL);
  cvt(ioW,   Wio,  DMODEL*DMODEL);
  cvt(kvW,   Wkv,  2*DMODEL*DMODEL);

  const float scale = 0.125f;          // 1/sqrt(64)
  const long HB = (long)HEAD;          // per-head elem stride
  const long SB = (long)DEC * DEC;     // per-head score stride

  // r_head_k = r @ rnet_w^T -> gbuf [1024][512] (TDM-staged GEMM)
  gemm_bf16_t<4,4,true><<<dim3(8, 4, 1), 128, 0, stream>>>(Rb, Wrn, DEC, DMODEL, DMODEL,
      0, 0, 0, gbuf, nullptr, nullptr, 0, nullptr, 1.f, 0, MODE_PLAIN);
  split_r_kernel<<<dim3(DEC*DMODEL/256), 256, 0, stream>>>(gbuf, rh);

  // w_heads = w @ qkv_w^T -> gbuf [8192][1536], then split per head (+biases)
  gemm_bf16_t<4,4,true><<<dim3(64, 12, 1), 128, 0, stream>>>(Xw, Wqkv, DEC*BSZ, 3*DMODEL, DMODEL,
      0, 0, 0, gbuf, nullptr, nullptr, 0, nullptr, 1.f, 0, MODE_PLAIN);
  split_qkv_kernel<<<dim3(DEC*BSZ*DMODEL/256), 256, 0, stream>>>(gbuf, rwb, rrb, Qrw, Qrr, Kh, VT);

  // ---- self attention, per batch index (heads batched on grid.z) ----
  for (int b = 0; b < BSZ; ++b) {
    const size_t ho = (size_t)b * NHEADS * HEAD;
    // BD_raw[n] = Q_rr @ rh[n]^T   (1024x1024, K=64)
    gemm_bf16_t<4,4,false><<<dim3(8, 8, NHEADS), 128, 0, stream>>>(Qrr + ho, rh, DEC, DEC, DHEAD,
        HB, HB, SB, BDf, nullptr, nullptr, 0, nullptr, 1.f, b, MODE_PLAIN);
    // S = (Q_rw @ K^T + shifted BD) * scale, causal masked
    gemm_bf16_t<4,4,false><<<dim3(8, 8, NHEADS), 128, 0, stream>>>(Qrw + ho, Kh + ho, DEC, DEC, DHEAD,
        HB, HB, SB, Sf, nullptr, BDf, SB, nullptr, scale, b, MODE_SELF_SCORE);
    softmax_rows<<<dim3(NHEADS*DEC/8), 256, 0, stream>>>(Sf, Ph, DEC);
    // attn_vec = P @ V -> scatter bf16 into AV[(i*BSZ+b)*DMODEL + n*64 + d]
    gemm_bf16_t<2,2,false><<<dim3(16, 1, NHEADS), 128, 0, stream>>>(Ph, VT + ho, DEC, DHEAD, DEC,
        SB, HB, 0, nullptr, AV, nullptr, 0, nullptr, 1.f, b, MODE_HEADVEC);
  }

  // attn_out = attn_vec @ o_w^T -> gbuf ; ln1 = LN(w + attn_out)
  gemm_bf16_t<4,4,true><<<dim3(64, 4, 1), 128, 0, stream>>>(AV, Wo, DEC*BSZ, DMODEL, DMODEL,
      0, 0, 0, gbuf, nullptr, nullptr, 0, nullptr, 1.f, 0, MODE_PLAIN);
  add_layernorm_kernel<<<dim3(DEC*BSZ/8), 256, 0, stream>>>(w_f, gbuf, ln1g, ln1b, ln1f, ln1h);

  // ---- cross attention ----
  // inter_q = ln1 @ q_w^T, scattered directly into per-head layout (reuse Qrw)
  gemm_bf16_t<4,4,true><<<dim3(64, 4, 1), 128, 0, stream>>>(ln1h, Wq, DEC*BSZ, DMODEL, DMODEL,
      0, 0, 0, nullptr, Qrw, nullptr, 0, nullptr, 1.f, 0, MODE_SPLIT_HEAD);
  // inter_kv = enc @ kv_w^T -> gbuf [8192][1024], split per head (reuse Kh/VT)
  gemm_bf16_t<4,4,true><<<dim3(64, 8, 1), 128, 0, stream>>>(Xe, Wkv, ENCL*BSZ, 2*DMODEL, DMODEL,
      0, 0, 0, gbuf, nullptr, nullptr, 0, nullptr, 1.f, 0, MODE_PLAIN);
  split_kv_kernel<<<dim3(ENCL*BSZ*DMODEL/256), 256, 0, stream>>>(gbuf, Kh, VT);

  for (int b = 0; b < BSZ; ++b) {
    const size_t ho = (size_t)b * NHEADS * HEAD;
    gemm_bf16_t<4,4,false><<<dim3(8, 8, NHEADS), 128, 0, stream>>>(Qrw + ho, Kh + ho, DEC, ENCL, DHEAD,
        HB, HB, SB, Sf, nullptr, nullptr, 0, pad, scale, b, MODE_CROSS_SCORE);
    softmax_rows<<<dim3(NHEADS*DEC/8), 256, 0, stream>>>(Sf, Ph, ENCL);
    gemm_bf16_t<2,2,false><<<dim3(16, 1, NHEADS), 128, 0, stream>>>(Ph, VT + ho, DEC, DHEAD, ENCL,
        SB, HB, 0, nullptr, XV, nullptr, 0, nullptr, 1.f, b, MODE_HEADVEC);
  }

  // x_out = x_vec @ io_w^T -> gbuf ; out = LN(x_out + ln1)
  gemm_bf16_t<4,4,true><<<dim3(64, 4, 1), 128, 0, stream>>>(XV, Wio, DEC*BSZ, DMODEL, DMODEL,
      0, 0, 0, gbuf, nullptr, nullptr, 0, nullptr, 1.f, 0, MODE_PLAIN);
  add_layernorm_kernel<<<dim3(DEC*BSZ/8), 256, 0, stream>>>(gbuf, ln1f, ln2g, ln2b,
      (float*)d_out, nullptr);
}
